// Model_7945689497774
// MI455X (gfx1250) — matrix operand order repaired and checked by reading, not compiled, not run on hardware
//
#include <hip/hip_runtime.h>
#include <hip/hip_bf16.h>

// ---------------------------------------------------------------------------
// Problem constants (from the reference)
// ---------------------------------------------------------------------------
#define NN 20000
#define EE 320000
#define DD 1024
#define GG 64

typedef __attribute__((ext_vector_type(16))) __bf16 v16bf;
typedef __attribute__((ext_vector_type(8)))  __bf16 v8bf;
typedef __attribute__((ext_vector_type(4)))  __bf16 v4bf;
typedef __attribute__((ext_vector_type(8)))  float  v8f;
typedef __attribute__((ext_vector_type(4)))  int    v4i;

#if __has_builtin(__builtin_amdgcn_global_load_async_to_lds_b128)
#define HAVE_ASYNC_LDS 1
#endif

// builtin-compatible address-space-qualified pointer types
typedef __attribute__((address_space(1))) v4i* gv4i_p;
typedef __attribute__((address_space(3))) v4i* lv4i_p;

// ---------------------------------------------------------------------------
// 0) init: deg = 1 (self loop), sums = 0, cnt = 0
// ---------------------------------------------------------------------------
__global__ __launch_bounds__(256) void k_init(float* __restrict__ deg,
                                              float* __restrict__ sums,
                                              float* __restrict__ cnt) {
  int i = blockIdx.x * 256 + threadIdx.x;     // launched with GG*DD threads
  if (i < NN) deg[i] = 1.0f;
  if (i < GG * DD) sums[i] = 0.0f;
  if (i < GG) cnt[i] = 0.0f;
}

// ---------------------------------------------------------------------------
// 1) degree: deg[dst] += 1
// ---------------------------------------------------------------------------
__global__ __launch_bounds__(256) void k_deg(const long long* __restrict__ ei,
                                             float* __restrict__ deg) {
  int e = blockIdx.x * 256 + threadIdx.x;
  if (e < EE) atomicAdd(&deg[(int)ei[EE + e]], 1.0f);
}

// ---------------------------------------------------------------------------
// 2) dinv = rsqrt(deg); also per-graph node counts
// ---------------------------------------------------------------------------
__global__ __launch_bounds__(256) void k_dinv(const float* __restrict__ deg,
                                              float* __restrict__ dinv,
                                              const long long* __restrict__ batch,
                                              float* __restrict__ cnt) {
  int n = blockIdx.x * 256 + threadIdx.x;
  if (n < NN) {
    dinv[n] = rsqrtf(deg[n]);
    atomicAdd(&cnt[(int)batch[n]], 1.0f);
  }
}

// ---------------------------------------------------------------------------
// 2b) split W into bf16 hi/lo planes once (removes per-iteration VALU from GEMM)
// ---------------------------------------------------------------------------
__global__ __launch_bounds__(256) void k_cvtw(const float* __restrict__ W,
                                              __bf16* __restrict__ whi,
                                              __bf16* __restrict__ wlo) {
  int i = (blockIdx.x * 256 + threadIdx.x) * 4;     // D*D = 1,048,576 elements
  const float4 v = *(const float4*)(W + i);
  float f[4] = {v.x, v.y, v.z, v.w};
  v4bf h, l;
#pragma unroll
  for (int j = 0; j < 4; ++j) {
    __bf16 hh = (__bf16)f[j];
    h[j] = hh;
    l[j] = (__bf16)(f[j] - (float)hh);
  }
  *(v4bf*)(whi + i) = h;
  *(v4bf*)(wlo + i) = l;
}

// ---------------------------------------------------------------------------
// 3) WMMA GEMM:  y[n,:] = (x[n,:] @ W2^T) * dinv[n]  (also written to agg as
//    the self-loop init).  Split-bf16 (hi*hi + hi*lo + lo*hi) for ~fp32
//    fidelity on the matrix pipe.  B tiles (shared by all 8 waves) are staged
//    in LDS via async global->LDS copies; A is converted in-registers.
//    Wave tile: 16 rows x 64 cols.  Block: 8 waves = 128 rows x 64 cols.
// ---------------------------------------------------------------------------
#define BROW 40   // padded LDS row stride in bf16 (80B: 16B-aligned, conflict-spread)

__global__ __launch_bounds__(256) void k_gemm(const float* __restrict__ x,
                                              const __bf16* __restrict__ whi,
                                              const __bf16* __restrict__ wlo,
                                              const float* __restrict__ dinv,
                                              float* __restrict__ y,
                                              float* __restrict__ agg) {
  __shared__ __bf16 lds_hi[64 * BROW];
  __shared__ __bf16 lds_lo[64 * BROW];

  const int tid  = threadIdx.x;
  const int lane = tid & 31;
  const int wave = tid >> 5;
  const int m16  = lane & 15;
  const int hi   = lane >> 4;                    // 0 or 1 (lane half)

  int rowBase = blockIdx.x * 128 + wave * 16;
  const bool valid = (rowBase < NN);
  if (!valid) rowBase = 0;                       // keep wave alive for barriers
  const int colBase = blockIdx.y * 64;

  const float* __restrict__ xrow = x + (long long)(rowBase + m16) * DD;

  // staging assignment: thread -> (tile row tr, 16B quarter tq)
  const int tr = tid >> 2;                       // 0..63
  const int tq = tid & 3;                        // 0..3  (8 bf16 each)
  const __bf16* ghi = whi + (long long)(colBase + tr) * DD + tq * 8;
  const __bf16* glo = wlo + (long long)(colBase + tr) * DD + tq * 8;
  __bf16* lhi = lds_hi + tr * BROW + tq * 8;
  __bf16* llo = lds_lo + tr * BROW + tq * 8;

  v8f acc[4] = {v8f{}, v8f{}, v8f{}, v8f{}};

  for (int k0 = 0; k0 < DD; k0 += 32) {
    __syncthreads();                             // LDS reads of prev iter done
#if HAVE_ASYNC_LDS
    __builtin_amdgcn_global_load_async_to_lds_b128((gv4i_p)(ghi + k0), (lv4i_p)lhi, 0, 0);
    __builtin_amdgcn_global_load_async_to_lds_b128((gv4i_p)(glo + k0), (lv4i_p)llo, 0, 0);
#if __has_builtin(__builtin_amdgcn_s_wait_asynccnt)
    __builtin_amdgcn_s_wait_asynccnt(0);
#else
    asm volatile("s_wait_asynccnt 0x0" ::: "memory");
#endif
#else
    *(v8bf*)lhi = *(const v8bf*)(ghi + k0);
    *(v8bf*)llo = *(const v8bf*)(glo + k0);
#endif
    __syncthreads();                             // tile visible to all waves

    // ---- A fragment (16x32 bf16 lane layout), convert fp32 -> hi/lo ----
    __builtin_prefetch(xrow + k0 + 32, 0, 1);
    const float4 a0 = *(const float4*)(xrow + k0 + 8 * hi);
    const float4 a1 = *(const float4*)(xrow + k0 + 8 * hi + 4);
    const float4 a2 = *(const float4*)(xrow + k0 + 8 * hi + 16);
    const float4 a3 = *(const float4*)(xrow + k0 + 8 * hi + 20);
    float af[16] = {a0.x, a0.y, a0.z, a0.w, a1.x, a1.y, a1.z, a1.w,
                    a2.x, a2.y, a2.z, a2.w, a3.x, a3.y, a3.z, a3.w};
    v16bf a_hi, a_lo;
#pragma unroll
    for (int j = 0; j < 16; ++j) {
      __bf16 h = (__bf16)af[j];
      a_hi[j] = h;
      a_lo[j] = (__bf16)(af[j] - (float)h);
    }

    // ---- 4 B tiles from LDS (2 x ds_load_b128 per plane), 3 WMMAs each ----
#pragma unroll
    for (int t = 0; t < 4; ++t) {
      const __bf16* bh = lds_hi + (t * 16 + m16) * BROW + 8 * hi;
      const __bf16* bl = lds_lo + (t * 16 + m16) * BROW + 8 * hi;
      v8bf bh0 = *(const v8bf*)(bh);
      v8bf bh1 = *(const v8bf*)(bh + 16);
      v8bf bl0 = *(const v8bf*)(bl);
      v8bf bl1 = *(const v8bf*)(bl + 16);
      v16bf b_hi = __builtin_shufflevector(bh0, bh1, 0, 1, 2, 3, 4, 5, 6, 7,
                                           8, 9, 10, 11, 12, 13, 14, 15);
      v16bf b_lo = __builtin_shufflevector(bl0, bl1, 0, 1, 2, 3, 4, 5, 6, 7,
                                           8, 9, 10, 11, 12, 13, 14, 15);
      v8f c = acc[t];
      c = __builtin_amdgcn_wmma_f32_16x16x32_bf16(false, a_hi, false, b_hi,
                                                  (short)0, c, false, false);
      c = __builtin_amdgcn_wmma_f32_16x16x32_bf16(false, a_hi, false, b_lo,
                                                  (short)0, c, false, false);
      c = __builtin_amdgcn_wmma_f32_16x16x32_bf16(false, a_lo, false, b_hi,
                                                  (short)0, c, false, false);
      acc[t] = c;
    }
  }

  if (!valid) return;

  // ---- store: scale by dinv[row]; write y and agg (self-loop init) ----
  float dv[8];
#pragma unroll
  for (int r = 0; r < 8; ++r) dv[r] = dinv[rowBase + 8 * hi + r];
#pragma unroll
  for (int t = 0; t < 4; ++t) {
    int col = colBase + t * 16 + m16;
#pragma unroll
    for (int r = 0; r < 8; ++r) {
      int orow = rowBase + 8 * hi + r;
      float v = acc[t][r] * dv[r];
      long long idx = (long long)orow * DD + col;
      y[idx]   = v;
      agg[idx] = v;
    }
  }
}

// ---------------------------------------------------------------------------
// 4) edge scatter: agg[dst,:] += y[src,:]   (one block per edge, float4/lane)
// ---------------------------------------------------------------------------
__global__ __launch_bounds__(256) void k_edge(const long long* __restrict__ ei,
                                              const float* __restrict__ y,
                                              float* __restrict__ agg) {
  int e   = blockIdx.x;
  int src = (int)ei[e];
  int dst = (int)ei[EE + e];
  int d   = threadIdx.x * 4;
  const float4 v = *(const float4*)(y + (long long)src * DD + d);
  float* p = agg + (long long)dst * DD + d;
  atomicAdd(p + 0, v.x);
  atomicAdd(p + 1, v.y);
  atomicAdd(p + 2, v.z);
  atomicAdd(p + 3, v.w);
}

// ---------------------------------------------------------------------------
// 5) pool: h = relu(dinv[n]*agg[n,:] + b2); sums[batch[n],:] += h
//    batch is sorted -> accumulate 64 consecutive nodes in registers, flush
//    atomics only at graph boundaries (g is block-uniform each iteration).
// ---------------------------------------------------------------------------
__global__ __launch_bounds__(256) void k_pool(const float* __restrict__ agg,
                                              const float* __restrict__ dinv,
                                              const long long* __restrict__ batch,
                                              const float* __restrict__ b2,
                                              float* __restrict__ sums) {
  const int t  = threadIdx.x;
  const int d  = t * 4;
  const int n0 = blockIdx.x * 64;
  const float4 bb = *(const float4*)(b2 + d);
  float ax = 0.f, ay = 0.f, az = 0.f, aw = 0.f;
  int curg = -1;
  for (int i = 0; i < 64; ++i) {
    int n = n0 + i;
    if (n >= NN) break;
    int g = (int)batch[n];
    if (g != curg) {
      if (curg >= 0) {
        float* p = sums + curg * DD + d;
        atomicAdd(p + 0, ax); atomicAdd(p + 1, ay);
        atomicAdd(p + 2, az); atomicAdd(p + 3, aw);
      }
      curg = g; ax = ay = az = aw = 0.f;
    }
    float s = dinv[n];
    const float4 a = *(const float4*)(agg + (long long)n * DD + d);
    ax += fmaxf(fmaf(s, a.x, bb.x), 0.f);
    ay += fmaxf(fmaf(s, a.y, bb.y), 0.f);
    az += fmaxf(fmaf(s, a.z, bb.z), 0.f);
    aw += fmaxf(fmaf(s, a.w, bb.w), 0.f);
  }
  if (curg >= 0) {
    float* p = sums + curg * DD + d;
    atomicAdd(p + 0, ax); atomicAdd(p + 1, ay);
    atomicAdd(p + 2, az); atomicAdd(p + 3, aw);
  }
}

// ---------------------------------------------------------------------------
// 6) head: out[g,c] = dot(sums[g,:], Wr[c,:]) / max(cnt[g],1) + br[c]
// ---------------------------------------------------------------------------
__global__ __launch_bounds__(128) void k_final(const float* __restrict__ sums,
                                               const float* __restrict__ cnt,
                                               const float* __restrict__ Wr,
                                               const float* __restrict__ br,
                                               float* __restrict__ out) {
  int t = threadIdx.x;
  if (t >= 2 * GG) return;
  int g = t >> 1, c = t & 1;
  float acc = 0.f;
  for (int d = 0; d < DD; ++d) acc = fmaf(sums[g * DD + d], Wr[c * DD + d], acc);
  out[g * 2 + c] = acc / fmaxf(cnt[g], 1.0f) + br[c];
}

// ---------------------------------------------------------------------------
// launcher
// ---------------------------------------------------------------------------
extern "C" void kernel_launch(void* const* d_in, const int* in_sizes, int n_in,
                              void* d_out, int out_size, void* d_ws, size_t ws_size,
                              hipStream_t stream) {
  const float*     x     = (const float*)d_in[0];
  const long long* ei    = (const long long*)d_in[1];   // [2,E] int64: src row 0, dst row 1
  const long long* batch = (const long long*)d_in[2];
  const float*     W2    = (const float*)d_in[7];       // only the last block survives
  const float*     b2    = (const float*)d_in[8];
  const float*     Wr    = (const float*)d_in[9];
  const float*     br    = (const float*)d_in[10];
  float*           out   = (float*)d_out;

  // workspace carve-up (256B aligned)
  char* ws = (char*)d_ws;
  size_t off = 0;
  auto take = [&](size_t bytes) {
    char* p = ws + off;
    off += (bytes + 255) & ~(size_t)255;
    return p;
  };
  float*  y    = (float*)take((size_t)NN * DD * 4);   // xw * dinv[row]
  float*  agg  = (float*)take((size_t)NN * DD * 4);   // scatter accumulator
  float*  deg  = (float*)take((size_t)NN * 4);
  float*  dinv = (float*)take((size_t)NN * 4);
  float*  sums = (float*)take((size_t)GG * DD * 4);
  float*  cnt  = (float*)take((size_t)GG * 4);
  __bf16* whi  = (__bf16*)take((size_t)DD * DD * 2);
  __bf16* wlo  = (__bf16*)take((size_t)DD * DD * 2);
  (void)ws_size; (void)in_sizes; (void)n_in; (void)out_size;

  // 0) init
  k_init<<<(GG * DD + 255) / 256, 256, 0, stream>>>(deg, sums, cnt);
  // 1) degrees
  k_deg<<<(EE + 255) / 256, 256, 0, stream>>>(ei, deg);
  // 2) dinv + counts
  k_dinv<<<(NN + 255) / 256, 256, 0, stream>>>(deg, dinv, batch, cnt);
  // 2b) W -> bf16 hi/lo planes
  k_cvtw<<<(DD * DD / 4) / 256, 256, 0, stream>>>(W2, whi, wlo);
  // 3) WMMA GEMM: 128-row x 64-col blocks
  dim3 gg((NN + 127) / 128, DD / 64);
  k_gemm<<<gg, 256, 0, stream>>>(x, whi, wlo, dinv, y, agg);
  // 4) edge scatter
  k_edge<<<EE, 256, 0, stream>>>(ei, y, agg);
  // 5) relu + segment-sum pooling
  k_pool<<<(NN + 63) / 64, 256, 0, stream>>>(agg, dinv, batch, b2, sums);
  // 6) head
  k_final<<<1, 128, 0, stream>>>(sums, cnt, Wr, br, out);
}
